// ResGraphConv_13589276524722
// MI455X (gfx1250) — compile-verified, matrix operand order checked by reading, not run on hardware
//
#include <hip/hip_runtime.h>

// ResGraphConv on MI455X (gfx1250, wave32, WMMA).
// Per layer: dual-GEMM (f16-in/f32-acc v_wmma, weights staged fragment-major in
// LDS, bias pre-splatted into the C accumulator, fully-unrolled N-tile loop
// with immediate-offset stores) -> zero nsum -> edge scatter (coalesced 512B
// gathers + f32 atomics) -> relu combine. Final: (x + h2)*0.5.

typedef _Float16 v16h __attribute__((ext_vector_type(16)));
typedef _Float16 v8h  __attribute__((ext_vector_type(8)));
typedef float    v8f  __attribute__((ext_vector_type(8)));

#define DFEAT 128

// Build a 16-half A/B fragment from 32 consecutive-K f32 values:
// per lane, K runs [base..base+7] and [base+16..base+23] (ISA 7.12.2, 16-bit 16x32 A).
__device__ __forceinline__ v16h frag_from_f32(const float* __restrict__ p) {
  float4 a0 = *(const float4*)(p);
  float4 a1 = *(const float4*)(p + 4);
  float4 b0 = *(const float4*)(p + 16);
  float4 b1 = *(const float4*)(p + 20);
  v16h r;
  r[0]  = (_Float16)a0.x; r[1]  = (_Float16)a0.y; r[2]  = (_Float16)a0.z; r[3]  = (_Float16)a0.w;
  r[4]  = (_Float16)a1.x; r[5]  = (_Float16)a1.y; r[6]  = (_Float16)a1.z; r[7]  = (_Float16)a1.w;
  r[8]  = (_Float16)b0.x; r[9]  = (_Float16)b0.y; r[10] = (_Float16)b0.z; r[11] = (_Float16)b0.w;
  r[12] = (_Float16)b1.x; r[13] = (_Float16)b1.y; r[14] = (_Float16)b1.z; r[15] = (_Float16)b1.w;
  return r;
}

__device__ __forceinline__ v8f splat8(float v) {
  v8f r;
#pragma unroll
  for (int j = 0; j < 8; ++j) r[j] = v;
  return r;
}

// Fused dual GEMM: y0 = x*w0^T + b0, y1 = x*w1^T + b1.
// Both 128x128 weight matrices are converted f32->f16 and staged in LDS in
// fragment-major layout: slot = mat*1024 + nt*128 + kt*32 + lane, 16 halves
// (32B) per slot, 64KB total. Inner loop reads are pure ds_load_b128 pairs;
// bias lives in the accumulator init; stores use one base + immediate offsets.
__global__ __launch_bounds__(256) void gemm_dual(const float* __restrict__ x,
                                                 const float* __restrict__ w0,
                                                 const float* __restrict__ w1,
                                                 const float* __restrict__ b0,
                                                 const float* __restrict__ b1,
                                                 float* __restrict__ y0,
                                                 float* __restrict__ y1,
                                                 int nrows) {
  __shared__ _Float16 wfrag[2 * 8 * 4 * 32 * 16];   // 32768 halves = 64KB

  const int lane = threadIdx.x & 31;
  const int wave = threadIdx.x >> 5;
  const int m    = lane & 15;   // M (or N for B-frags)
  const int hl   = lane >> 4;   // lane-half selects K sub-run

  // ---- Stage both weight matrices into LDS (all 256 threads; 8 slots each).
#pragma unroll
  for (int s = 0; s < 8; ++s) {
    const int slot  = (int)threadIdx.x + s * 256;    // 0..2047
    const int lane2 = slot & 31;
    const int kt2   = (slot >> 5) & 3;
    const int nt2   = (slot >> 7) & 7;
    const int mat   = slot >> 10;
    const int row   = nt2 * 16 + (lane2 & 15);       // B[k][n] = w[n][k]
    const int kb    = kt2 * 32 + (lane2 >> 4) * 8;
    const float* p  = (mat ? w1 : w0) + row * DFEAT + kb;
    v16h f = frag_from_f32(p);
    _Float16* q = wfrag + (size_t)slot * 16;
    *(v8h*)q       = *(v8h*)&f;
    *(v8h*)(q + 8) = *((v8h*)&f + 1);
  }
  __syncthreads();

  const long rowBase = (long)blockIdx.x * 256 + (long)wave * 32;
  if (rowBase >= nrows) return;  // uniform per wave; V % 32 == 0 so strips never straddle

  // Pre-load per-lane biases for all 8 N-tiles (col = nt*16 + m fixed per lane).
  float pb0[8], pb1[8];
#pragma unroll
  for (int nt = 0; nt < 8; ++nt) {
    pb0[nt] = b0[nt * 16 + m];
    pb1[nt] = b1[nt * 16 + m];
  }

  // A fragments for both strips, all K tiles (resident: 64 VGPRs).
  v16h a[2][4];
#pragma unroll
  for (int s = 0; s < 2; ++s) {
    const float* xr = x + (rowBase + s * 16 + m) * DFEAT + hl * 8;
#pragma unroll
    for (int kt = 0; kt < 4; ++kt) a[s][kt] = frag_from_f32(xr + kt * 32);
  }

  // Single base address per output; all store offsets are compile-time imms.
  float* __restrict__ py0 = y0 + rowBase * DFEAT + (long)hl * 8 * DFEAT + m;
  float* __restrict__ py1 = y1 + rowBase * DFEAT + (long)hl * 8 * DFEAT + m;

#pragma unroll
  for (int nt = 0; nt < 8; ++nt) {
    v8f c00 = splat8(pb0[nt]), c01 = splat8(pb0[nt]);
    v8f c10 = splat8(pb1[nt]), c11 = splat8(pb1[nt]);
    const _Float16* base0 = wfrag + (size_t)(nt * 128 + lane) * 16;
#pragma unroll
    for (int kt = 0; kt < 4; ++kt) {
      v16h f0 = *(const v16h*)(base0 + (size_t)(kt * 32) * 16);
      v16h f1 = *(const v16h*)(base0 + (size_t)(1024 + kt * 32) * 16);
      c00 = __builtin_amdgcn_wmma_f32_16x16x32_f16(false, a[0][kt], false, f0, (short)0, c00, false, false);
      c01 = __builtin_amdgcn_wmma_f32_16x16x32_f16(false, a[1][kt], false, f0, (short)0, c01, false, false);
      c10 = __builtin_amdgcn_wmma_f32_16x16x32_f16(false, a[0][kt], false, f1, (short)0, c10, false, false);
      c11 = __builtin_amdgcn_wmma_f32_16x16x32_f16(false, a[1][kt], false, f1, (short)0, c11, false, false);
    }
#pragma unroll
    for (int r = 0; r < 8; ++r) {   // C layout: row = base + hl*8 + r, col = lane%16
      py0[nt * 16 + r * DFEAT]                 = c00[r];
      py0[nt * 16 + r * DFEAT + 16 * DFEAT]    = c01[r];
      py1[nt * 16 + r * DFEAT]                 = c10[r];
      py1[nt * 16 + r * DFEAT + 16 * DFEAT]    = c11[r];
    }
  }
}

// One wave per edge: lanes cover 128 columns as float4 => coalesced 512B row
// gathers and contiguous global_atomic_add_f32 bursts, both directions.
__global__ __launch_bounds__(256) void scatter_edges(const float* __restrict__ y1,
                                                     const int* __restrict__ edges,
                                                     float* __restrict__ nsum,
                                                     int numEdges) {
  long t = (long)blockIdx.x * blockDim.x + threadIdx.x;
  int e = (int)(t >> 5);
  if (e >= numEdges) return;
  int c = ((int)t & 31) * 4;
  long src = edges[2 * e];
  long dst = edges[2 * e + 1];
  float4 vs = *(const float4*)(y1 + src * DFEAT + c);
  float4 vd = *(const float4*)(y1 + dst * DFEAT + c);
  float* pd = nsum + dst * DFEAT + c;
  float* ps = nsum + src * DFEAT + c;
  atomicAdd(pd + 0, vs.x); atomicAdd(pd + 1, vs.y); atomicAdd(pd + 2, vs.z); atomicAdd(pd + 3, vs.w);
  atomicAdd(ps + 0, vd.x); atomicAdd(ps + 1, vd.y); atomicAdd(ps + 2, vd.z); atomicAdd(ps + 3, vd.w);
}

__global__ __launch_bounds__(256) void relu_combine(const float* __restrict__ y0,
                                                    const float* __restrict__ ns,
                                                    float* __restrict__ h, long n4) {
  long i = (long)blockIdx.x * blockDim.x + threadIdx.x;
  if (i >= n4) return;
  i *= 4;
  float4 a = *(const float4*)(y0 + i);
  float4 b = *(const float4*)(ns + i);
  float4 o;
  o.x = fmaxf(a.x + b.x, 0.0f);
  o.y = fmaxf(a.y + b.y, 0.0f);
  o.z = fmaxf(a.z + b.z, 0.0f);
  o.w = fmaxf(a.w + b.w, 0.0f);
  *(float4*)(h + i) = o;
}

__global__ __launch_bounds__(256) void final_combine(const float* __restrict__ x,
                                                     const float* __restrict__ y0,
                                                     const float* __restrict__ ns,
                                                     float* __restrict__ out, long n4) {
  long i = (long)blockIdx.x * blockDim.x + threadIdx.x;
  if (i >= n4) return;
  i *= 4;
  float4 xi = *(const float4*)(x + i);
  float4 a  = *(const float4*)(y0 + i);
  float4 b  = *(const float4*)(ns + i);
  float4 o;
  o.x = (xi.x + fmaxf(a.x + b.x, 0.0f)) * 0.5f;
  o.y = (xi.y + fmaxf(a.y + b.y, 0.0f)) * 0.5f;
  o.z = (xi.z + fmaxf(a.z + b.z, 0.0f)) * 0.5f;
  o.w = (xi.w + fmaxf(a.w + b.w, 0.0f)) * 0.5f;
  *(float4*)(out + i) = o;
}

extern "C" void kernel_launch(void* const* d_in, const int* in_sizes, int n_in,
                              void* d_out, int out_size, void* d_ws, size_t ws_size,
                              hipStream_t stream) {
  const float* x    = (const float*)d_in[0];
  const int*   eidx = (const int*)d_in[1];
  const float* w0_1 = (const float*)d_in[2];
  const float* b0_1 = (const float*)d_in[3];
  const float* w1_1 = (const float*)d_in[4];
  const float* b1_1 = (const float*)d_in[5];
  const float* w0_2 = (const float*)d_in[6];
  const float* b0_2 = (const float*)d_in[7];
  const float* w1_2 = (const float*)d_in[8];
  const float* b1_2 = (const float*)d_in[9];

  const int  V  = in_sizes[0] / DFEAT;   // 300000
  const int  E  = in_sizes[1] / 2;       // 900000
  const long VD = (long)V * DFEAT;

  // Workspace: three V*D f32 buffers.
  float* bufA = (float*)d_ws;
  float* bufB = bufA + VD;
  float* bufC = bufB + VD;

  const dim3 gGemm((V + 255) / 256);
  const long scatterThreads = (long)E * 32;
  const dim3 gScat((unsigned)((scatterThreads + 255) / 256));
  const long n4 = VD / 4;
  const dim3 gElt((unsigned)((n4 + 255) / 256));

  // Layer 1: y0->A, y1->B, nsum->C, h->B
  gemm_dual<<<gGemm, 256, 0, stream>>>(x, w0_1, w1_1, b0_1, b1_1, bufA, bufB, V);
  hipMemsetAsync(bufC, 0, VD * sizeof(float), stream);
  scatter_edges<<<gScat, 256, 0, stream>>>(bufB, eidx, bufC, E);
  relu_combine<<<gElt, 256, 0, stream>>>(bufA, bufC, bufB, n4);

  // Layer 2: input h=B, y0->A, y1->C, nsum->B (B is free once gemm completes)
  gemm_dual<<<gGemm, 256, 0, stream>>>(bufB, w0_2, w1_2, b0_2, b1_2, bufA, bufC, V);
  hipMemsetAsync(bufB, 0, VD * sizeof(float), stream);
  scatter_edges<<<gScat, 256, 0, stream>>>(bufC, eidx, bufB, E);
  final_combine<<<gElt, 256, 0, stream>>>(x, bufA, bufB, (float*)d_out, n4);
}
